// FCOS_57647051047630
// MI455X (gfx1250) — matrix operand order repaired and checked by reading, compiled, not run on hardware
//
#include <hip/hip_runtime.h>

// ---------- types ----------
typedef __attribute__((ext_vector_type(8)))  __bf16 v8bf;
typedef __attribute__((ext_vector_type(16))) __bf16 v16bf;
typedef __attribute__((ext_vector_type(8)))  float  v8f;
typedef __attribute__((ext_vector_type(8)))  unsigned short ushort8;
typedef __attribute__((ext_vector_type(4)))  unsigned int v4u;
typedef __attribute__((ext_vector_type(8)))  int v8i;
typedef __attribute__((ext_vector_type(4)))  int v4i;

#define CIN    256
#define KTOT   2304   // 9 taps * 256 channels (tap-major)
#define NBATCH 2
#define MT     64     // block M tile (output channels)
#define NT     128    // block N tile (spatial positions)

#if defined(__has_builtin)
#if __has_builtin(__builtin_amdgcn_tensor_load_to_lds)
#define HAVE_TDM 1
#endif
#endif

static __device__ __forceinline__ unsigned short f2bf(float f) {
    union { float f; unsigned int u; } v; v.f = f;
    unsigned int u = v.u;
    unsigned int r = u + 0x7FFFu + ((u >> 16) & 1u);
    return (unsigned short)(r >> 16);
}

// ---------- fp32 -> bf16 elementwise ----------
__global__ __launch_bounds__(256)
void cvt_f32_to_bf16(const float* __restrict__ src, unsigned short* __restrict__ dst,
                     long long nElem) {
    long long i = (long long)blockIdx.x * blockDim.x + threadIdx.x;
    long long stride = (long long)gridDim.x * blockDim.x;
    for (; i < nElem; i += stride) dst[i] = f2bf(src[i]);
}

// ---------- weight pack: [Cout,Cin,3,3] f32 -> [Cout, tap*256+chan] bf16 ----------
__global__ __launch_bounds__(256)
void pack_w_bf16(const float* __restrict__ src, unsigned short* __restrict__ dst,
                 int Cout) {
    const long long total = (long long)Cout * KTOT;
    long long i = (long long)blockIdx.x * blockDim.x + threadIdx.x;
    long long stride = (long long)gridDim.x * blockDim.x;
    for (; i < total; i += stride) {
        int m = (int)(i / KTOT);
        int k = (int)(i - (long long)m * KTOT);
        int r = k >> 8;
        int c = k & 255;
        dst[i] = f2bf(src[((long long)m * 256 + c) * 9 + r]);
    }
}

// ---------- implicit-GEMM conv3x3 (SAME), bf16 WMMA, fp32 accumulate ----------
__global__ __launch_bounds__(256)
void conv3x3_bf16_wmma(const unsigned short* __restrict__ xbf,
                       const unsigned short* __restrict__ wbf,
                       const float* __restrict__ bias,
                       float* __restrict__ out,
                       int Cout, int H, int W, int logW,
                       long long nStride, long long cStride,
                       int chBase, long long pOff, int reluFlag) {
    __shared__ unsigned short __attribute__((aligned(64))) ldsA[MT * 32]; // [Mrow][K]
    __shared__ unsigned short __attribute__((aligned(64))) ldsB[NT * 32]; // [Ncol][K]
    __shared__ float          __attribute__((aligned(64))) ldsBias[MT];

    const int tid   = threadIdx.x;
    const int lane  = tid & 31;
    const int wave  = tid >> 5;
    const int waveM = (wave & 1) << 5;    // 0,32
    const int waveN = (wave >> 1) << 5;   // 0,32,64,96
    const int n     = blockIdx.z;
    const int m0    = blockIdx.y * MT;
    const int pBase = blockIdx.x * NT;
    const int HW    = H << logW;
    const long long xN = (long long)n * CIN * HW;

    // ---- bias -> LDS via async copy (16 lanes x 16B = 64 floats) ----
    if (tid < 16) {
        const int bi = m0 + (tid << 2);
        const float* bsrc = bias + ((bi + 3 < Cout) ? bi : 0); // clamped rows are store-guarded
        const unsigned bl = (unsigned)(size_t)&ldsBias[tid << 2];
        asm volatile("global_load_async_to_lds_b128 %0, %1, off"
                     :: "v"(bl), "v"(bsrc) : "memory");
    }

    // ---- B staging: thread owns column bj, 16 channels starting at bkk ----
    const int bj  = tid & 127;
    const int bkk = (tid >> 7) << 4;      // 0 or 16
    const int p   = pBase + bj;
    const int py  = p >> logW;
    const int px  = p & (W - 1);

    v8f acc[2][2];
#pragma unroll
    for (int i = 0; i < 2; ++i)
#pragma unroll
        for (int j = 0; j < 2; ++j)
            acc[i][j] = (v8f){0.f,0.f,0.f,0.f,0.f,0.f,0.f,0.f};

    // ---- A (weights) tile source: advances 64B per K-step, contiguous over the loop ----
    unsigned long long aAddr =
        (unsigned long long)(size_t)wbf + (unsigned long long)m0 * (KTOT * 2);
    const unsigned ldsAaddr = (unsigned)(size_t)&ldsA[0];

#ifdef HAVE_TDM
    // D# group1: data_size=8B, tensor row=576 units, rows=Cout-m0 (OOB rows -> 0),
    // tile = 8 units x 64 rows, row stride = 576 units.
    v8i g1;
    g1[0] = 0x30000;                          // workgroup_mask=0 | data_size=3 (8B)
    g1[1] = (int)(576u << 16);                // tensor_dim0 low16
    g1[2] = (int)((unsigned)(Cout - m0) << 16); // tensor_dim1 low16
    g1[3] = (int)(8u << 16);                  // tile_dim0 = 8
    g1[4] = 64;                               // tile_dim1 = 64, tile_dim2 = 0
    g1[5] = 576;                              // tensor_dim0_stride low32
    g1[6] = 0;
    g1[7] = 0;
    const v4i gz = {0, 0, 0, 0};
#ifdef __HIP_HAS_GFX1250_TDM_HEADER_PLACEHOLDER__
#endif
#else
    // fallback: per-thread async b128 (256 * 16B = 4KB tile)
    const int aRow = tid >> 2;
    const int aSeg = (tid & 3) << 3;
    const int aM   = m0 + aRow;
    const unsigned aLds = (unsigned)(size_t)&ldsA[(aRow << 5) + aSeg];
    const unsigned short* aG =
        wbf + (long long)(aM < Cout ? aM : 0) * KTOT + aSeg;
#endif

    for (int tap = 0; tap < 9; ++tap) {
        const int t3 = tap / 3;
        const int dy = t3 - 1;
        const int dx = tap - t3 * 3 - 1;
        const int yy = py + dy, xx = px + dx;
        const bool valid = (yy >= 0) && (yy < H) && (xx >= 0) && (xx < W);
        const unsigned short* src = xbf + xN + (long long)bkk * HW
            + ((long long)(valid ? yy : 0) << logW) + (valid ? xx : 0);

        for (int cc = 0; cc < CIN; cc += 32) {
            __syncthreads();

#ifdef HAVE_TDM
            if (wave == 0) {
                v4u g0;
                g0[0] = 1u;                                  // count=1
                g0[1] = ldsAaddr;                            // lds_addr
                g0[2] = (unsigned)aAddr;                     // global_addr[31:0]
                g0[3] = (unsigned)((aAddr >> 32) & 0x01FFFFFFull) | (2u << 30); // type=2
#if __has_include(<hip/amd_detail/amd_gfx1250_TDM.h>)
                const v8i gz8 = {0,0,0,0,0,0,0,0};
                __builtin_amdgcn_tensor_load_to_lds(g0, g1, gz, gz, gz8, 0);
#else
                __builtin_amdgcn_tensor_load_to_lds(g0, g1, gz, gz, 0);
#endif
            }
#else
            asm volatile("global_load_async_to_lds_b128 %0, %1, off"
                         :: "v"(aLds), "v"(aG) : "memory");
#endif

            // B tile: single divergent guard around all 16 channel-gathers
            ushort8 b0 = {0,0,0,0,0,0,0,0};
            ushort8 b1 = {0,0,0,0,0,0,0,0};
            if (valid) {
#pragma unroll
                for (int q = 0; q < 8; ++q) b0[q] = src[q * HW];
#pragma unroll
                for (int q = 0; q < 8; ++q) b1[q] = src[(q + 8) * HW];
            }
            *(ushort8*)&ldsB[(bj << 5) + bkk]     = b0;
            *(ushort8*)&ldsB[(bj << 5) + bkk + 8] = b1;

#ifdef HAVE_TDM
            __builtin_amdgcn_s_wait_tensorcnt(0);
#else
            asm volatile("s_wait_asynccnt 0x0" ::: "memory");
#endif
            __syncthreads();

            // ---- fragments + 4 WMMA ----
            const int mr   = lane & 15;
            const int kseg = (lane >> 4) << 4;
            v16bf aF[2], bF[2];
#pragma unroll
            for (int i = 0; i < 2; ++i) {
                const unsigned short* pa = &ldsA[((waveM + (i << 4) + mr) << 5) + kseg];
                v8bf lo = *(const v8bf*)pa;
                v8bf hi = *(const v8bf*)(pa + 8);
                aF[i] = __builtin_shufflevector(lo, hi,
                                                0,1,2,3,4,5,6,7,8,9,10,11,12,13,14,15);
                const unsigned short* pb =
                    &ldsB[((waveN + (i << 4) + mr) << 5) + ((lane >> 4) << 4)];
                bF[i] = *(const v16bf*)pb;
            }
#pragma unroll
            for (int i = 0; i < 2; ++i)
#pragma unroll
                for (int j = 0; j < 2; ++j)
                    acc[i][j] = __builtin_amdgcn_wmma_f32_16x16x32_bf16(
                        false, aF[i], false, bF[j], (short)0, acc[i][j], false, false);

            aAddr += 64;                 // next 32 halfs of the weight row block
            src   += (long long)32 * HW; // next 32 input channels
#ifndef HAVE_TDM
            aG    += 32;
#endif
        }
    }

    // bias became visible long ago; make it formal: issuing wave waits, then barrier
    asm volatile("s_wait_asynccnt 0x0" ::: "memory");
    __syncthreads();

    // ---- store: VGPR q -> M = q + 8*(lane>=16), N = lane&15 ----
    const int mrBase = waveM + ((lane >> 4) << 3);
#pragma unroll
    for (int i = 0; i < 2; ++i) {
#pragma unroll
        for (int j = 0; j < 2; ++j) {
            const int nIdx = pBase + waveN + (j << 4) + (lane & 15);
            if (nIdx < HW) {
#pragma unroll
                for (int q = 0; q < 8; ++q) {
                    const int ml = mrBase + (i << 4) + q;   // local row
                    const int m  = m0 + ml;
                    if (m < Cout) {
                        float v = acc[i][j][q] + ldsBias[ml];
                        if (reluFlag) v = v > 0.f ? v : 0.f;
                        out[(long long)n * nStride + (long long)(chBase + m) * cStride
                            + pOff + nIdx] = v;
                    }
                }
            }
        }
    }
}

// ---------- GroupNorm(16) + ReLU, fp32 in -> bf16 out ----------
__global__ __launch_bounds__(256)
void groupnorm_relu_bf16(const float* __restrict__ x,
                         const float* __restrict__ gamma,
                         const float* __restrict__ beta,
                         unsigned short* __restrict__ out,
                         int HW, int logHW) {
    __shared__ float ssum[256];
    __shared__ float ssq[256];
    __shared__ float sstat[2];
    const int tid = threadIdx.x;
    const int n   = blockIdx.x >> 4;
    const int g   = blockIdx.x & 15;
    const long long base = ((long long)n * 256 + (g << 4)) * HW;
    const int len = HW << 4;

    float s = 0.f, q = 0.f;
    for (int i = tid; i < len; i += 256) {
        float v = x[base + i];
        s += v; q += v * v;
    }
    ssum[tid] = s; ssq[tid] = q;
    __syncthreads();
    for (int off = 128; off > 0; off >>= 1) {
        if (tid < off) { ssum[tid] += ssum[tid + off]; ssq[tid] += ssq[tid + off]; }
        __syncthreads();
    }
    if (tid == 0) {
        float mean = ssum[0] / (float)len;
        float var  = ssq[0] / (float)len - mean * mean;
        sstat[0] = mean;
        sstat[1] = rsqrtf(var + 1e-5f);
    }
    __syncthreads();
    const float mean = sstat[0], inv = sstat[1];
    for (int i = tid; i < len; i += 256) {
        int c = (g << 4) + (i >> logHW);
        float v = (x[base + i] - mean) * inv * gamma[c] + beta[c];
        v = v > 0.f ? v : 0.f;
        out[base + i] = f2bf(v);
    }
}

// ------------------------------------------------------------------
extern "C" void kernel_launch(void* const* d_in, const int* in_sizes, int n_in,
                              void* d_out, int out_size, void* d_ws, size_t ws_size,
                              hipStream_t stream) {
    (void)in_sizes; (void)n_in; (void)out_size; (void)ws_size;

    const float* feat[5] = {(const float*)d_in[0], (const float*)d_in[1],
                            (const float*)d_in[2], (const float*)d_in[3],
                            (const float*)d_in[4]};
    const float* clsOutW = (const float*)d_in[29];
    const float* clsOutB = (const float*)d_in[30];
    const float* regOutW = (const float*)d_in[31];
    const float* regOutB = (const float*)d_in[32];
    const float* ctrW    = (const float*)d_in[33];
    const float* ctrB    = (const float*)d_in[34];

    char*  ws  = (char*)d_ws;
    size_t cur = 0;
    auto alloc = [&](size_t bytes) -> void* {
        void* p = ws + cur;
        cur = (cur + bytes + 255) & ~(size_t)255;
        return p;
    };
    const size_t TW = (size_t)256 * KTOT;
    unsigned short* wClsT[3], *wRegT[3];
    for (int i = 0; i < 3; ++i) wClsT[i] = (unsigned short*)alloc(TW * 2);
    for (int i = 0; i < 3; ++i) wRegT[i] = (unsigned short*)alloc(TW * 2);
    unsigned short* wClsOut = (unsigned short*)alloc((size_t)80 * KTOT * 2);
    unsigned short* wRegOut = (unsigned short*)alloc((size_t)4  * KTOT * 2);
    unsigned short* wCtr    = (unsigned short*)alloc((size_t)1  * KTOT * 2);

    const size_t actMax = (size_t)NBATCH * 256 * 16384;
    unsigned short* actA = (unsigned short*)alloc(actMax * 2);
    unsigned short* actB = (unsigned short*)alloc(actMax * 2);
    unsigned short* actC = (unsigned short*)alloc(actMax * 2);
    float*          convF = (float*)alloc(actMax * 4);

    for (int i = 0; i < 3; ++i)
        pack_w_bf16<<<512, 256, 0, stream>>>((const float*)d_in[5 + 4 * i],  wClsT[i], 256);
    for (int i = 0; i < 3; ++i)
        pack_w_bf16<<<512, 256, 0, stream>>>((const float*)d_in[17 + 4 * i], wRegT[i], 256);
    pack_w_bf16<<<256, 256, 0, stream>>>(clsOutW, wClsOut, 80);
    pack_w_bf16<<<32,  256, 0, stream>>>(regOutW, wRegOut, 4);
    pack_w_bf16<<<8,   256, 0, stream>>>(ctrW,    wCtr,    1);

    const int  Hs[5]    = {128, 64, 32, 16, 8};
    const int  logWs[5] = {7, 6, 5, 4, 3};
    const long long pOffs[5] = {0, 16384, 20480, 21504, 21760};
    const long long TOTAL = 21824;
    float* outF = (float*)d_out;

    for (int lvl = 0; lvl < 5; ++lvl) {
        const int H = Hs[lvl], W = Hs[lvl], logW = logWs[lvl];
        const int HW = H * W, logHW = 2 * logW;
        const int gN = (HW + NT - 1) / NT;

        cvt_f32_to_bf16<<<2048, 256, 0, stream>>>(feat[lvl], actA,
                                                  (long long)NBATCH * 256 * HW);

        for (int t = 0; t < 2; ++t) {
            const int pbase = t ? 17 : 5;
            const unsigned short* curp = actA;
            unsigned short* pp[2] = {actB, actC};
            for (int i = 0; i < 3; ++i) {
                const unsigned short* wb = t ? wRegT[i] : wClsT[i];
                const float* bias = (const float*)d_in[pbase + 4 * i + 1];
                dim3 grid(gN, 256 / MT, NBATCH);
                conv3x3_bf16_wmma<<<grid, 256, 0, stream>>>(
                    curp, wb, bias, convF, 256, H, W, logW,
                    (long long)256 * HW, (long long)HW, 0, 0, 0);
                const float* gm = (const float*)d_in[pbase + 4 * i + 2];
                const float* bt = (const float*)d_in[pbase + 4 * i + 3];
                groupnorm_relu_bf16<<<NBATCH * 16, 256, 0, stream>>>(
                    convF, gm, bt, pp[i & 1], HW, logHW);
                curp = pp[i & 1];
            }
            if (t == 0) {
                dim3 g(gN, (80 + MT - 1) / MT, NBATCH);
                conv3x3_bf16_wmma<<<g, 256, 0, stream>>>(
                    curp, wClsOut, clsOutB, outF, 80, H, W, logW,
                    85LL * TOTAL, TOTAL, 0, pOffs[lvl], 0);
            } else {
                dim3 g(gN, 1, NBATCH);
                conv3x3_bf16_wmma<<<g, 256, 0, stream>>>(
                    curp, wRegOut, regOutB, outF, 4, H, W, logW,
                    85LL * TOTAL, TOTAL, 80, pOffs[lvl], 1);
                conv3x3_bf16_wmma<<<g, 256, 0, stream>>>(
                    curp, wCtr, ctrB, outF, 1, H, W, logW,
                    85LL * TOTAL, TOTAL, 84, pOffs[lvl], 0);
            }
        }
    }
}
